// GRU_43722767073693
// MI455X (gfx1250) — matrix operand order repaired and checked by reading, not compiled, not run on hardware
//
#include <hip/hip_runtime.h>
#include <hip/hip_bf16.h>
#include <stdint.h>

// Problem dims
#define TT 4096
#define BB 32
#define II 256
#define HH 256
#define KK 512              // I + H
#define MM (TT*BB)          // 131072 rows
#define MT 64               // rows per workgroup tile
#define PLANE_DW 65536      // (KK*HH)/2 packed dwords per weight plane

typedef __attribute__((ext_vector_type(16))) _Float16 v16h;
typedef __attribute__((ext_vector_type(8)))  _Float16 v8h;
typedef __attribute__((ext_vector_type(4)))  _Float16 v4h;
typedef __attribute__((ext_vector_type(8)))  float    v8f;

__device__ __forceinline__ v8f wmma16(v16h a, v16h b, v8f c) {
  // D = A(16x32 f16) * B(32x16 f16) + C(16x16 f32)
  return __builtin_amdgcn_wmma_f32_16x16x32_f16(false, a, false, b, (short)0, c,
                                                false, false);
}

// LDS swizzle: row-major 512 halfs/row, 8-half (16B) blocks XORed by row to
// kill bank conflicts on the A-fragment reads (16 consecutive rows, same k run).
__device__ __forceinline__ int aswz(int row, int k) {
  return (row << 9) + ((((k >> 3) ^ (row & 15)) << 3) | (k & 7));
}

__device__ __forceinline__ float sigmoid_f(float x) { return 1.f / (1.f + __expf(-x)); }
__device__ __forceinline__ float tanh_f(float x)    { return 1.f - 2.f / (__expf(2.f * x) + 1.f); }

// ---------------------------------------------------------------------------
// Weight prep: fp32 W[l][k][n] -> hi/lo f16 planes, pre-swizzled into the
// per-lane B-fragment layout this code feeds V_WMMA_F32_16X16X32_F16 (contiguous k, as A):
//   plane dword index e = ((c*16 + nt)*32 + lane)*8 + j
//   holds packed { W[k0], W[k0+1] },  k0 = c*32 + (lane>=16)*16 + 2j,
//   n = nt*16 + (lane&15).
// ---------------------------------------------------------------------------
__global__ __launch_bounds__(256) void prep_weights_kernel(
    const float* __restrict__ Wr, const float* __restrict__ Wz,
    const float* __restrict__ Ws, uint32_t* __restrict__ wsw) {
  const int idx = blockIdx.x * 256 + threadIdx.x;   // 0 .. 2*3*65536-1
  if (idx >= 2 * 3 * PLANE_DW) return;
  const int l   = idx / (3 * PLANE_DW);
  const int rem = idx % (3 * PLANE_DW);
  const int g   = rem / PLANE_DW;                   // 0=r, 1=z, 2=s
  const int e   = rem % PLANE_DW;
  const int j    = e & 7;
  const int lane = (e >> 3) & 31;
  const int nt   = (e >> 8) & 15;
  const int c    = e >> 12;
  const int k0 = c * 32 + ((lane >> 4) << 4) + 2 * j;
  const int n  = nt * 16 + (lane & 15);
  const float* W  = (g == 0) ? Wr : (g == 1) ? Wz : Ws;
  const float* Wl = W + (size_t)l * KK * HH;
  const float w0 = Wl[(size_t)k0 * HH + n];
  const float w1 = Wl[(size_t)(k0 + 1) * HH + n];
  const _Float16 h0 = (_Float16)w0, h1 = (_Float16)w1;
  const _Float16 l0 = (_Float16)(w0 - (float)h0), l1 = (_Float16)(w1 - (float)h1);
  union { _Float16 h[2]; uint32_t u; } ph, pl;
  ph.h[0] = h0; ph.h[1] = h1;
  pl.h[0] = l0; pl.h[1] = l1;
  uint32_t* hiP = wsw + ((size_t)(l * 3 + g) * 2 + 0) * PLANE_DW;
  uint32_t* loP = wsw + ((size_t)(l * 3 + g) * 2 + 1) * PLANE_DW;
  hiP[e] = ph.u;
  loP[e] = pl.u;
}

// ---------------------------------------------------------------------------
// One GRU "layer" pass over MT=64 rows per workgroup.
// 256 threads = 8 waves. Wave w owns ALL 4 m-subtiles and n-tiles {2w, 2w+1},
// so every weight B-fragment is loaded from global exactly once per workgroup
// (no cross-wave duplication); A-fragments are re-read from LDS (cheap).
// Split-precision GEMM: ah*bh + al*bh + ah*bl  (fp32-accurate).
// ---------------------------------------------------------------------------
__global__ __launch_bounds__(256) void gru_layer_kernel(
    const float* __restrict__ inps, const float* __restrict__ h0,
    const uint32_t* __restrict__ wsw, int layer,
    const float* __restrict__ br_l, const float* __restrict__ bz_l,
    const float* __restrict__ bs_l, float* __restrict__ out,
    long long row_base_arg, long long out_base_row, long long out_valid) {
  extern __shared__ __align__(16) _Float16 smem[];
  _Float16* sAhi = smem;                 // MT x 512 halfs (swizzled)
  _Float16* sAlo = smem + MT * KK;

  const long long row_base = row_base_arg + (long long)blockIdx.x * MT;
  const int tid  = threadIdx.x;
  const int lane = tid & 31;
  const int w    = tid >> 5;
  const int ntg0 = w << 1;               // this wave's first n-tile (of 16)
  const int lh8  = (lane >> 4) << 4;     // 0 or 16 (A-frag K sub-run select)

  // ---- Stage A = [x | h] into LDS as hi/lo f16 planes ------------------
  auto store4 = [&](int row, int k, float4 f) {
    const int a = aswz(row, k);
    const _Float16 a0 = (_Float16)f.x, a1 = (_Float16)f.y;
    const _Float16 a2 = (_Float16)f.z, a3 = (_Float16)f.w;
    v4h hv = {a0, a1, a2, a3};
    v4h lv = {(_Float16)(f.x - (float)a0), (_Float16)(f.y - (float)a1),
              (_Float16)(f.z - (float)a2), (_Float16)(f.w - (float)a3)};
    *(v4h*)&sAhi[a] = hv;
    *(v4h*)&sAlo[a] = lv;
  };
  for (int idx = tid; idx < (MT * II) / 4; idx += 256) {
    const int row = idx >> 6;
    const int k   = (idx & 63) << 2;
    const float4 v = *(const float4*)(inps + (size_t)(row_base + row) * II + k);
    store4(row, k, v);
  }
  for (int idx = tid; idx < (MT * HH) / 4; idx += 256) {
    const int row = idx >> 6;                 // row_base % 32 == 0 -> b = row&31
    const int k   = (idx & 63) << 2;
    const float4 v = *(const float4*)(h0 + (size_t)(row & 31) * HH + k);
    store4(row, 256 + k, v);
  }
  __syncthreads();

  // ---- split-precision GEMM over K=512 ---------------------------------
  // acc[ms*2 + ntl] is the 16x16 tile (m-subtile ms, n-tile ntg0+ntl)
  auto run_gemm = [&](const uint32_t* __restrict__ wH,
                      const uint32_t* __restrict__ wL, v8f acc[8]) {
#pragma unroll 1
    for (int c = 0; c < 16; ++c) {
      const int kb = c << 5;
      const size_t off0 = ((((size_t)c * 16) + ntg0 + 0) * 32 + lane) << 3;
      const size_t off1 = ((((size_t)c * 16) + ntg0 + 1) * 32 + lane) << 3;
      v16h bh0 = *(const v16h*)(wH + off0);
      v16h bl0 = *(const v16h*)(wL + off0);
      v16h bh1 = *(const v16h*)(wH + off1);
      v16h bl1 = *(const v16h*)(wL + off1);
#pragma unroll
      for (int ms = 0; ms < 4; ++ms) {
        const int arow = (ms << 4) + (lane & 15);
        v8h xh0 = *(const v8h*)&sAhi[aswz(arow, kb + lh8)];
        v8h xh1 = *(const v8h*)&sAhi[aswz(arow, kb + 8 + lh8)];
        v8h xl0 = *(const v8h*)&sAlo[aswz(arow, kb + lh8)];
        v8h xl1 = *(const v8h*)&sAlo[aswz(arow, kb + 8 + lh8)];
        v16h ah = __builtin_shufflevector(xh0, xh1, 0,1,2,3,4,5,6,7,8,9,10,11,12,13,14,15);
        v16h al = __builtin_shufflevector(xl0, xl1, 0,1,2,3,4,5,6,7,8,9,10,11,12,13,14,15);
        acc[ms * 2 + 0] = wmma16(ah, bh0, acc[ms * 2 + 0]);
        acc[ms * 2 + 0] = wmma16(al, bh0, acc[ms * 2 + 0]);
        acc[ms * 2 + 0] = wmma16(ah, bl0, acc[ms * 2 + 0]);
        acc[ms * 2 + 1] = wmma16(ah, bh1, acc[ms * 2 + 1]);
        acc[ms * 2 + 1] = wmma16(al, bh1, acc[ms * 2 + 1]);
        acc[ms * 2 + 1] = wmma16(ah, bl1, acc[ms * 2 + 1]);
      }
    }
  };

  const uint32_t* wrH = wsw + ((size_t)(layer * 3 + 0) * 2 + 0) * PLANE_DW;
  const uint32_t* wrL = wsw + ((size_t)(layer * 3 + 0) * 2 + 1) * PLANE_DW;
  const uint32_t* wzH = wsw + ((size_t)(layer * 3 + 1) * 2 + 0) * PLANE_DW;
  const uint32_t* wzL = wsw + ((size_t)(layer * 3 + 1) * 2 + 1) * PLANE_DW;
  const uint32_t* wsH = wsw + ((size_t)(layer * 3 + 2) * 2 + 0) * PLANE_DW;
  const uint32_t* wsL = wsw + ((size_t)(layer * 3 + 2) * 2 + 1) * PLANE_DW;

  v8f acc[8];
  v8f zs[8];

  // C/D layout: VGPR v, lanes 0-15 -> M=v, N=lane; lanes 16-31 -> M=8+v, N=lane-16.
  const int nlo   = lane & 15;
  const int mhalf = (lane >> 4) << 3;

  // ---- gate z ----------------------------------------------------------
#pragma unroll
  for (int i = 0; i < 8; ++i) { v8f zz = {}; acc[i] = zz; }
  run_gemm(wzH, wzL, acc);
#pragma unroll
  for (int ms = 0; ms < 4; ++ms)
#pragma unroll
    for (int ntl = 0; ntl < 2; ++ntl) {
      const int t = ms * 2 + ntl;
      const int n = ((ntg0 + ntl) << 4) + nlo;
      const float bias = bz_l[n];
#pragma unroll
      for (int v = 0; v < 8; ++v) zs[t][v] = sigmoid_f(acc[t][v] + bias);
    }

  // ---- gate r, then overwrite LDS h-part with r*h ----------------------
#pragma unroll
  for (int i = 0; i < 8; ++i) { v8f zz = {}; acc[i] = zz; }
  run_gemm(wrH, wrL, acc);
  __syncthreads();   // all waves done reading original h-part
#pragma unroll
  for (int ms = 0; ms < 4; ++ms)
#pragma unroll
    for (int ntl = 0; ntl < 2; ++ntl) {
      const int t = ms * 2 + ntl;
      const int n = ((ntg0 + ntl) << 4) + nlo;
      const float bias = br_l[n];
#pragma unroll
      for (int v = 0; v < 8; ++v) {
        const int mloc = (ms << 4) + mhalf + v;
        const float hv = h0[(size_t)(mloc & 31) * HH + n];
        const float rh = sigmoid_f(acc[t][v] + bias) * hv;
        const _Float16 hi = (_Float16)rh;
        sAhi[aswz(mloc, 256 + n)] = hi;
        sAlo[aswz(mloc, 256 + n)] = (_Float16)(rh - (float)hi);
      }
    }
  __syncthreads();

  // ---- gate s, combine, store -----------------------------------------
#pragma unroll
  for (int i = 0; i < 8; ++i) { v8f zz = {}; acc[i] = zz; }
  run_gemm(wsH, wsL, acc);
#pragma unroll
  for (int ms = 0; ms < 4; ++ms)
#pragma unroll
    for (int ntl = 0; ntl < 2; ++ntl) {
      const int t = ms * 2 + ntl;
      const int n = ((ntg0 + ntl) << 4) + nlo;
      const float bias = bs_l[n];
#pragma unroll
      for (int v = 0; v < 8; ++v) {
        const int mloc = (ms << 4) + mhalf + v;
        const float hv = h0[(size_t)(mloc & 31) * HH + n];
        const float sv = tanh_f(acc[t][v] + bias);
        const float zv = zs[t][v];
        const long long o = row_base + mloc - out_base_row;
        if (o >= 0 && o < out_valid)
          out[(size_t)o * HH + n] = zv * hv + (1.f - zv) * sv;
      }
    }
}

// ---------------------------------------------------------------------------
extern "C" void kernel_launch(void* const* d_in, const int* in_sizes, int n_in,
                              void* d_out, int out_size, void* d_ws, size_t ws_size,
                              hipStream_t stream) {
  (void)in_sizes; (void)n_in; (void)out_size; (void)ws_size;
  const float* inps = (const float*)d_in[0];
  const float* hidd = (const float*)d_in[1];
  const float* Wr   = (const float*)d_in[2];
  const float* b_r  = (const float*)d_in[3];
  const float* Wz   = (const float*)d_in[4];
  const float* b_z  = (const float*)d_in[5];
  const float* Ws   = (const float*)d_in[6];
  const float* b_s  = (const float*)d_in[7];
  float*    out = (float*)d_out;
  uint32_t* wsw = (uint32_t*)d_ws;

  const size_t lds_bytes = (size_t)MT * KK * 2 * sizeof(_Float16);  // 128 KB
  (void)hipFuncSetAttribute((const void*)gru_layer_kernel,
                            hipFuncAttributeMaxDynamicSharedMemorySize,
                            (int)lds_bytes);

  // 1) pack weights into hi/lo f16 B-fragment planes (3.0 MB in d_ws)
  prep_weights_kernel<<<(2 * 3 * PLANE_DW) / 256, 256, 0, stream>>>(Wr, Wz, Ws, wsw);

  // 2) layer 1 over all rows -> outs (T,B,H)
  gru_layer_kernel<<<MM / MT, 256, lds_bytes, stream>>>(
      inps, hidd + (size_t)1 * BB * HH, wsw, 1,
      b_r + HH, b_z + HH, b_s + HH,
      out, 0LL, 0LL, (long long)MM);

  // 3) layer 0: only t = T-1 matters -> hidds[0]
  const long long tail_base = (long long)MM - MT;        // 131008 (mult of 32)
  gru_layer_kernel<<<1, 256, lds_bytes, stream>>>(
      inps, hidd, wsw, 0,
      b_r, b_z, b_s,
      out + (size_t)TT * BB * HH,                        // hidds[0] region
      tail_base, (long long)(TT - 1) * BB, (long long)BB);

  // 4) hidds[1] = outs[T-1]
  (void)hipMemcpyAsync(out + (size_t)TT * BB * HH + (size_t)BB * HH,
                       out + (size_t)(TT - 1) * BB * HH,
                       (size_t)BB * HH * sizeof(float),
                       hipMemcpyDeviceToDevice, stream);
}
